// GNN_31971736551536
// MI455X (gfx1250) — matrix operand order repaired and checked by reading, not compiled, not run on hardware
//
#include <hip/hip_runtime.h>
#include <hip/hip_bf16.h>

#define N_NODES 8192
#define N_FEAT  128
#define HIDDEN  128
#define N_EDGES 262144
#define ALPHA_C 0.5f
#define BN_EPS  1e-5f
#define MASK_WPR (N_NODES / 32)   // 256 mask words per row

typedef __bf16 bf16;
typedef __attribute__((ext_vector_type(16))) __bf16 v16bf;
typedef __attribute__((ext_vector_type(8)))  __bf16 v8bf;
typedef __attribute__((ext_vector_type(8)))  float  v8f;

// ---------------------------------------------------------------------------
// Zero the adjacency bitmask + scalar accumulators (ws is poisoned by harness)
// ---------------------------------------------------------------------------
__global__ void clear_ws_kernel(unsigned int* __restrict__ mask,
                                float* __restrict__ sumsq,
                                unsigned int* __restrict__ nnz) {
    unsigned int i = blockIdx.x * blockDim.x + threadIdx.x;
    if (i < (unsigned int)(N_NODES * MASK_WPR)) mask[i] = 0u;
    if (i == 0) { *sumsq = 0.0f; *nnz = 0u; }
}

// ---------------------------------------------------------------------------
// Scatter COO edges into a dense bitmask (A[i,j] in {0,1}; dups collapse)
// ---------------------------------------------------------------------------
__global__ void build_adj_kernel(const int* __restrict__ ei,
                                 unsigned int* __restrict__ mask) {
    int e = blockIdx.x * blockDim.x + threadIdx.x;
    if (e >= N_EDGES) return;
    int s = ei[e];
    int d = ei[N_EDGES + e];
    atomicOr(&mask[(size_t)s * MASK_WPR + (d >> 5)], 1u << (d & 31));
}

// ---------------------------------------------------------------------------
// X_hat = (BN(relu(x@W1+b1)))@W2 + b2 ; store bf16 ; accumulate sum(X_hat^2)
// One row per 128-thread block.
// ---------------------------------------------------------------------------
__global__ __launch_bounds__(128)
void mlp_kernel(const float* __restrict__ x,  const float* __restrict__ W1,
                const float* __restrict__ b1, const float* __restrict__ gamma,
                const float* __restrict__ beta, const float* __restrict__ rmean,
                const float* __restrict__ rvar, const float* __restrict__ W2,
                const float* __restrict__ b2, bf16* __restrict__ xhat,
                float* __restrict__ sumsq) {
    __shared__ float xr[N_FEAT];
    __shared__ float h[HIDDEN];
    const int row = blockIdx.x;
    const int j   = threadIdx.x;

    xr[j] = x[(size_t)row * N_FEAT + j];
    __syncthreads();

    float acc = b1[j];
#pragma unroll 8
    for (int k = 0; k < N_FEAT; ++k)
        acc = fmaf(xr[k], W1[k * HIDDEN + j], acc);
    acc = fmaxf(acc, 0.0f);                                   // ReLU
    acc = (acc - rmean[j]) * rsqrtf(rvar[j] + BN_EPS) * gamma[j] + beta[j];
    h[j] = acc;
    __syncthreads();

    float o = b2[j];
#pragma unroll 8
    for (int k = 0; k < HIDDEN; ++k)
        o = fmaf(h[k], W2[k * HIDDEN + j], o);

    xhat[(size_t)row * HIDDEN + j] = (bf16)o;

    float ss = o * o;                                         // Frobenius accum
    for (int off = 16; off > 0; off >>= 1) ss += __shfl_down(ss, off, 32);
    if ((threadIdx.x & 31) == 0) atomicAdd(sumsq, ss);
}

// ---------------------------------------------------------------------------
// Fused: S = Xhat@Xhat^T / sumsq ; Z = relu(S - prob + 0.5*A) * same_graph ;
// nnz count. 128x128 tile per 256-thread (8-wave) block, bf16 WMMA.
// ---------------------------------------------------------------------------
__global__ __launch_bounds__(256)
void fused_gemm_kernel(const bf16* __restrict__ xhat,
                       const unsigned int* __restrict__ mask,
                       const int* __restrict__ batch,
                       const float* __restrict__ sumsq,
                       const float* __restrict__ prob,
                       float* __restrict__ out,
                       unsigned int* __restrict__ nnz) {
    __shared__ __align__(64) bf16 ldsA[128 * HIDDEN];   // rows i0..i0+127
    __shared__ __align__(64) bf16 ldsB[128 * HIDDEN];   // rows j0..j0+127
    __shared__ unsigned int ldsMask[128 * 4];           // 128 rows x 4 words
    __shared__ int ldsBI[128];
    __shared__ int ldsBJ[128];

    const int tid = threadIdx.x;
    const int i0  = blockIdx.y * 128;
    const int j0  = blockIdx.x * 128;

    // Stage both Xhat row-blocks (contiguous 32 KB each) into LDS
    const uint4* srcA = (const uint4*)(xhat + (size_t)i0 * HIDDEN);
    const uint4* srcB = (const uint4*)(xhat + (size_t)j0 * HIDDEN);
    uint4* dA = (uint4*)ldsA;
    uint4* dB = (uint4*)ldsB;
#pragma unroll
    for (int it = 0; it < 8; ++it) {
        dA[tid + it * 256] = srcA[tid + it * 256];
        dB[tid + it * 256] = srcB[tid + it * 256];
    }
    // Stage adjacency bitmask slice: 512 words (rows i0..+127, cols j0..+127)
    {
        int t0 = tid, t1 = tid + 256;
        ldsMask[t0] = mask[(size_t)(i0 + (t0 >> 2)) * MASK_WPR + (j0 >> 5) + (t0 & 3)];
        ldsMask[t1] = mask[(size_t)(i0 + (t1 >> 2)) * MASK_WPR + (j0 >> 5) + (t1 & 3)];
    }
    if (tid < 128) { ldsBI[tid] = batch[i0 + tid]; ldsBJ[tid] = batch[j0 + tid]; }
    __syncthreads();

    const int lane    = tid & 31;
    const int half    = lane >> 4;        // 0: lanes 0-15, 1: lanes 16-31
    const int lr      = lane & 15;
    const int m_local = (tid >> 5) * 16;  // each wave owns a 16-row strip

    v8f acc[8];
#pragma unroll
    for (int t = 0; t < 8; ++t) { v8f z = {}; acc[t] = z; }

    // K = 128 -> 4 steps of K=32. ISA A-layout (16-bit 16x32): lanes 0-15 hold
    // K {0..7,16..23}, lanes 16-31 hold K {8..15,24..31} -> two contiguous
    // 16B chunks per lane. B-layout: lane n holds K {0..7,16..23} (half 0) or
    // {8..15,24..31} (half 1) of column n -> two contiguous 16B chunks per lane.
    const bf16* arow = ldsA + (m_local + lr) * HIDDEN;
#pragma unroll
    for (int kk = 0; kk < 4; ++kk) {
        const int kb = kk * 32;
        v8bf alo = *(const v8bf*)(arow + kb + half * 8);
        v8bf ahi = *(const v8bf*)(arow + kb + 16 + half * 8);
        v16bf afrag = __builtin_shufflevector(alo, ahi,
            0, 1, 2, 3, 4, 5, 6, 7, 8, 9, 10, 11, 12, 13, 14, 15);
#pragma unroll
        for (int t = 0; t < 8; ++t) {
            const bf16* brow = ldsB + (t * 16 + lr) * HIDDEN + kb + half * 8;
            v8bf blo = *(const v8bf*)brow;
            v8bf bhi = *(const v8bf*)(brow + 16);
            v16bf bfrag = __builtin_shufflevector(blo, bhi,
                0, 1, 2, 3, 4, 5, 6, 7, 8, 9, 10, 11, 12, 13, 14, 15);
            acc[t] = __builtin_amdgcn_wmma_f32_16x16x32_bf16(
                false, afrag, false, bfrag, (short)0, acc[t], false, false);
        }
    }

    // Epilogue. D-layout: lanes 0-15 -> M=r, N=lane; lanes 16-31 -> M=8+r.
    const float inv = 1.0f / sumsq[0];
    const float p   = prob[0];
    unsigned int cnt = 0;
#pragma unroll
    for (int t = 0; t < 8; ++t) {
        const int cl = t * 16 + lr;            // local column
        const int bj = ldsBJ[cl];
        const int wc = cl >> 5;
#pragma unroll
        for (int r = 0; r < 8; ++r) {
            const int rl = m_local + half * 8 + r;   // local row
            float s = acc[t][r] * inv;
            float a = ((ldsMask[rl * 4 + wc] >> (cl & 31)) & 1u) ? ALPHA_C : 0.0f;
            float z = fmaxf(s - p + a, 0.0f);
            if (ldsBI[rl] != bj) z = 0.0f;
            cnt += (z != 0.0f) ? 1u : 0u;
            __builtin_nontemporal_store(
                z, out + (size_t)(i0 + rl) * N_NODES + (j0 + cl));
        }
    }
    for (int off = 16; off > 0; off >>= 1) cnt += __shfl_down(cnt, off, 32);
    if (lane == 0) atomicAdd(nnz, cnt);
}

// ---------------------------------------------------------------------------
__global__ void finalize_kernel(const unsigned int* __restrict__ nnz,
                                const float* __restrict__ prob,
                                float* __restrict__ out) {
    out[(size_t)N_NODES * N_NODES]     = (float)(*nnz) / (float)N_EDGES;
    out[(size_t)N_NODES * N_NODES + 1] = prob[0];
}

// ---------------------------------------------------------------------------
extern "C" void kernel_launch(void* const* d_in, const int* in_sizes, int n_in,
                              void* d_out, int out_size, void* d_ws, size_t ws_size,
                              hipStream_t stream) {
    (void)in_sizes; (void)n_in; (void)out_size; (void)ws_size;
    const float* x     = (const float*)d_in[0];
    const int*   ei    = (const int*)d_in[1];
    const int*   batch = (const int*)d_in[2];
    const float* W1    = (const float*)d_in[3];
    const float* b1    = (const float*)d_in[4];
    const float* gamma = (const float*)d_in[5];
    const float* beta  = (const float*)d_in[6];
    const float* rmean = (const float*)d_in[7];
    const float* rvar  = (const float*)d_in[8];
    const float* W2    = (const float*)d_in[9];
    const float* b2    = (const float*)d_in[10];
    const float* prob  = (const float*)d_in[11];
    float* out = (float*)d_out;

    // Workspace layout: [8MB adjacency bitmask][2MB bf16 Xhat][scalars]
    char* ws = (char*)d_ws;
    unsigned int* mask  = (unsigned int*)ws;
    bf16*         xhat  = (bf16*)(ws + (size_t)8 * 1024 * 1024);
    float*        sumsq = (float*)(ws + (size_t)10 * 1024 * 1024);
    unsigned int* nnz   = (unsigned int*)(ws + (size_t)10 * 1024 * 1024 + 128);

    clear_ws_kernel<<<(N_NODES * MASK_WPR) / 256, 256, 0, stream>>>(mask, sumsq, nnz);
    build_adj_kernel<<<N_EDGES / 256, 256, 0, stream>>>(ei, mask);
    mlp_kernel<<<N_NODES, 128, 0, stream>>>(x, W1, b1, gamma, beta, rmean, rvar,
                                            W2, b2, xhat, sumsq);
    fused_gemm_kernel<<<dim3(N_NODES / 128, N_NODES / 128), 256, 0, stream>>>(
        xhat, mask, batch, sumsq, prob, out, nnz);
    finalize_kernel<<<1, 1, 0, stream>>>(nnz, prob, out);
}
